// MINI_DGCNN_MODULAR_21698174780228
// MI455X (gfx1250) — compile-verified, matrix operand order checked by reading, not run on hardware
//
#include <hip/hip_runtime.h>
#include <hip/hip_bf16.h>

typedef __attribute__((ext_vector_type(16))) _Float16 v16h;
typedef __attribute__((ext_vector_type(8)))  _Float16 v8h;
typedef __attribute__((ext_vector_type(8)))  float    v8f;
typedef _Float16 h16;

#define NBATCH 8
#define NPTS   2048
#define KNN_K  8

// ---- CDNA5 async global->LDS copy (ASYNCcnt-tracked DMA into LDS) ----
static __device__ __forceinline__ unsigned lds_off(const void* p) {
  // flat LDS pointer: low 32 bits are the LDS address (ISA 10.2 aperture rule)
  return (unsigned)(unsigned long long)p;
}
static __device__ __forceinline__ void async_g2l_b128(unsigned ldsaddr, const void* gaddr) {
  asm volatile("global_load_async_to_lds_b128 %0, %1, off"
               :: "v"(ldsaddr), "v"(gaddr) : "memory");
}
static __device__ __forceinline__ void wait_async0() {
  asm volatile("s_wait_asynccnt 0x0" ::: "memory");
}

// ===================== tiled WMMA GEMM =====================
// D[i,j] (+)= sum_k A[i,k]*B[k,j]; f16 in, f32 accum (v_wmma_f32_16x16x32_f16).
// 4 waves / block; block computes a 64x64 tile; each wave a 16x64 strip with
// 4 accumulators. AMODE=0: A K-contiguous (weights) -> async b128 global->LDS.
// AMODE=1: A M-contiguous (prev^T for KNN) -> vector load + LDS scatter.
// B (always N-contiguous) staged transposed [j][k]. Fragments: 2x ds_read_b128.
#define TM 64
#define TN 64
#define TK 32
#define LDP 40   // padded LDS row stride (halves); keeps 16B alignment

template <int AMODE>
__global__ __launch_bounds__(128) void gemm_wmma(
    const h16* __restrict__ A, long sa_i, long sa_k, long a_bs,
    const h16* __restrict__ Bm, long sb_k, long b_bs,
    void* __restrict__ Dp, int d_f16, long sd_i, long d_bs,
    int M, int Mrows, int Kd, int accum)
{
  __shared__ __align__(16) h16 lA[TM * LDP];
  __shared__ __align__(16) h16 lB[TN * LDP];   // holds B^T: [j][k]

  const int  j0 = blockIdx.x * TN;
  const int  i0 = blockIdx.y * TM;
  const long bz = blockIdx.z;
  A  += bz * a_bs;
  Bm += bz * b_bs;

  const int tid  = threadIdx.x;
  const int wave = tid >> 5;
  const int lane = tid & 31;
  const int half = lane >> 4;
  const int lidx = lane & 15;
  const int iw   = i0 + wave * 16;       // this wave's 16-row strip
  const bool aFull = (i0 + TM <= Mrows); // block-uniform

  v8f acc[4];
  if (accum && d_f16) {
    const h16* Dh = (const h16*)Dp + bz * d_bs;
#pragma unroll
    for (int jt = 0; jt < 4; ++jt) {
      const int j = j0 + jt * 16 + lidx;
#pragma unroll
      for (int v = 0; v < 8; ++v) {
        int i = iw + v + (half << 3);
        acc[jt][v] = (i < M) ? (float)Dh[(long)i * sd_i + j] : 0.f;
      }
    }
  } else {
#pragma unroll
    for (int jt = 0; jt < 4; ++jt)
#pragma unroll
      for (int v = 0; v < 8; ++v) acc[jt][v] = 0.f;
  }

  for (int kt = 0; kt < Kd; kt += TK) {
    // ---------- stage A tile: lA[i][k] = A[i0+i][kt+k] ----------
    if (AMODE == 0) {
      if (aFull) {
        // async DMA: global (K-contiguous) -> LDS, 16B per lane-chunk
#pragma unroll
        for (int r = 0; r < 2; ++r) {
          int c = tid + r * 128;            // 256 chunks of 8 halves
          int i = c >> 2, ks = (c & 3) * 8;
          const h16* ga = A + (long)(i0 + i) * sa_i + (kt + ks);
          async_g2l_b128(lds_off(&lA[i * LDP + ks]), ga);
          if (kt + TK < Kd) __builtin_prefetch(ga + TK, 0, 3);
        }
      } else {
#pragma unroll
        for (int r = 0; r < 2; ++r) {
          int c = tid + r * 128;
          int i = c >> 2, ks = (c & 3) * 8;
          v8h va = {};
          if (i0 + i < Mrows) va = *(const v8h*)(A + (long)(i0 + i) * sa_i + (kt + ks));
          *(v8h*)&lA[i * LDP + ks] = va;
        }
      }
    } else {
      // M contiguous (prev^T): 16B loads of 8 i's, transpose-scatter to [i][k]
#pragma unroll
      for (int r = 0; r < 2; ++r) {
        int c = tid + r * 128;
        int k = c >> 3, is = (c & 7) * 8;
        const h16* ga = A + (long)(i0 + is) + (long)(kt + k) * sa_k;
        v8h va = *(const v8h*)ga;
        if (kt + TK < Kd) __builtin_prefetch(ga + TK * sa_k, 0, 3);
#pragma unroll
        for (int u = 0; u < 8; ++u) lA[(is + u) * LDP + k] = va[u];
      }
    }
    // ---------- stage B tile (transposed): lB[j][k] = B[kt+k][j0+j] ----------
#pragma unroll
    for (int r = 0; r < 2; ++r) {
      int c = tid + r * 128;
      int k = c >> 3, js = (c & 7) * 8;
      const h16* gb = Bm + (long)(kt + k) * sb_k + (j0 + js);
      v8h vb = *(const v8h*)gb;
      if (kt + TK < Kd) __builtin_prefetch(gb + TK * sb_k, 0, 3);
#pragma unroll
      for (int u = 0; u < 8; ++u) lB[(js + u) * LDP + k] = vb[u];
    }
    if (AMODE == 0 && aFull) wait_async0();
    __syncthreads();

    // ---------- fragments + 4x WMMA ----------
    // 16-bit A 16x32 layout: lane<16 holds K {0..7,16..23}; lane>=16 holds
    // K {8..15,24..31}. Two aligned b128 LDS reads per fragment.
    {
      const int arow = (wave * 16 + lidx) * LDP;
      v8h alo = *(const v8h*)&lA[arow + half * 8];
      v8h ahi = *(const v8h*)&lA[arow + 16 + half * 8];
      v16h a = __builtin_shufflevector(alo, ahi, 0, 1, 2, 3, 4, 5, 6, 7,
                                       8, 9, 10, 11, 12, 13, 14, 15);
#pragma unroll
      for (int jt = 0; jt < 4; ++jt) {
        const int brow = (jt * 16 + lidx) * LDP;
        v8h blo = *(const v8h*)&lB[brow + half * 8];
        v8h bhi = *(const v8h*)&lB[brow + 16 + half * 8];
        v16h b = __builtin_shufflevector(blo, bhi, 0, 1, 2, 3, 4, 5, 6, 7,
                                         8, 9, 10, 11, 12, 13, 14, 15);
        acc[jt] = __builtin_amdgcn_wmma_f32_16x16x32_f16(
            false, a, false, b, (short)0, acc[jt], false, false);
      }
    }
    __syncthreads();
  }

  // ---------- store ----------
  const bool full = (i0 + TM <= M);      // wave-uniform fast path
  if (d_f16) {
    h16* Dh = (h16*)Dp + bz * d_bs;
#pragma unroll
    for (int jt = 0; jt < 4; ++jt) {
      const int j = j0 + jt * 16 + lidx;
#pragma unroll
      for (int v = 0; v < 8; ++v) {
        int i = iw + v + (half << 3);
        if (full || i < M) Dh[(long)i * sd_i + j] = (h16)acc[jt][v];
      }
    }
  } else {
    float* Df = (float*)Dp + bz * d_bs;
#pragma unroll
    for (int jt = 0; jt < 4; ++jt) {
      const int j = j0 + jt * 16 + lidx;
#pragma unroll
      for (int v = 0; v < 8; ++v) {
        int i = iw + v + (half << 3);
        if (full || i < M) Df[(long)i * sd_i + j] = acc[jt][v];
      }
    }
  }
}

// ===================== weight prep =====================
__global__ void conv_split(const float* __restrict__ W, h16* __restrict__ Wa,
                           h16* __restrict__ Wb, int O, int C)
{
  long e = (long)blockIdx.x * blockDim.x + threadIdx.x;
  if (e >= (long)O * C) return;
  int o = (int)(e / C), c = (int)(e % C);
  float w1 = W[(long)o * (2 * C) + c];
  float w2 = W[(long)o * (2 * C) + C + c];
  Wa[e] = (h16)w1;
  Wb[e] = (h16)(w2 - w1);
}

__global__ void conv_pad(const float* __restrict__ W, h16* __restrict__ Wh,
                         int rows_src, int cols, int rows_dst)
{
  long e = (long)blockIdx.x * blockDim.x + threadIdx.x;
  if (e >= (long)rows_dst * cols) return;
  int r = (int)(e / cols);
  Wh[e] = (r < rows_src) ? (h16)W[e] : (h16)0.f;
}

__global__ void zero_buf(float* p, int n)
{
  int e = blockIdx.x * blockDim.x + threadIdx.x;
  if (e < n) p[e] = 0.f;
}

// ===================== layer-0 P/Q (K=3, scalar) =====================
__global__ void pq0_kernel(const float* __restrict__ x, const float* __restrict__ W0,
                           h16* __restrict__ P, h16* __restrict__ Q)
{
  int n = blockIdx.x * blockDim.x + threadIdx.x;
  int o = blockIdx.y, b = blockIdx.z;
  const float* xp = x + ((long)b * NPTS + n) * 3;
  float p = 0.f, q = 0.f;
#pragma unroll
  for (int c = 0; c < 3; ++c) {
    float w1 = W0[o * 6 + c], w2 = W0[o * 6 + 3 + c];
    p += w1 * xp[c];
    q += (w2 - w1) * xp[c];
  }
  long off = ((long)b * 64 + o) * NPTS + n;
  P[off] = (h16)p;
  Q[off] = (h16)q;
}

// ===================== gather + max_k + BN stats =====================
__global__ void gather_max_stats(const h16* __restrict__ P, const h16* __restrict__ Q,
                                 const int* __restrict__ idx, h16* __restrict__ hmax,
                                 float* __restrict__ ssum, float* __restrict__ ssq,
                                 int O, int nchunk)
{
  int o = threadIdx.x, b = blockIdx.y, n0 = blockIdx.x * nchunk;
  const h16* Pb   = P + (long)b * O * NPTS + (long)o * NPTS;
  const h16* Qrow = Q + ((long)b * O + o) * NPTS;
  h16* Hrow       = hmax + ((long)b * O + o) * NPTS;
  const int* ib   = idx + (long)b * NPTS * KNN_K;
  float s = 0.f, s2 = 0.f;
  for (int n = n0; n < n0 + nchunk; ++n) {
    float q = (float)Qrow[n];
    float mx = -3.4e38f;
#pragma unroll
    for (int k = 0; k < KNN_K; ++k) {
      int j = ib[n * KNN_K + k];
      float h = (float)Pb[j] + q;
      s += h; s2 += h * h; mx = fmaxf(mx, h);
    }
    Hrow[n] = (h16)mx;
  }
  atomicAdd(&ssum[o], s);
  atomicAdd(&ssq[o], s2);
}

__global__ void chan_stats(const h16* __restrict__ Z, float* __restrict__ ssum,
                           float* __restrict__ ssq, int O, int nchunk)
{
  int o = threadIdx.x, b = blockIdx.y, n0 = blockIdx.x * nchunk;
  const h16* Zb = Z + ((long)b * O + o) * NPTS;
  float s = 0.f, s2 = 0.f;
  for (int n = n0; n < n0 + nchunk; ++n) {
    float h = (float)Zb[n];
    s += h; s2 += h * h;
  }
  atomicAdd(&ssum[o], s);
  atomicAdd(&ssq[o], s2);
}

__global__ void bn_lrelu(const h16* __restrict__ Hin, h16* __restrict__ Out,
                         const float* __restrict__ ssum, const float* __restrict__ ssq,
                         float inv_cnt, int O)
{
  long e = (long)blockIdx.x * blockDim.x + threadIdx.x;  // ((b*O+o)*NPTS+n)
  int o = (int)((e >> 11) & (O - 1));
  float m  = ssum[o] * inv_cnt;
  float var = ssq[o] * inv_cnt - m * m;
  float sc = rsqrtf(var + 1e-5f);
  float h = ((float)Hin[e] - m) * sc;
  Out[e] = (h16)(h > 0.f ? h : 0.2f * h);
}

// ===================== KNN =====================
__global__ void sqnorm(const h16* __restrict__ prev, float* __restrict__ xx, int C)
{
  int n = blockIdx.x * blockDim.x + threadIdx.x;
  int b = blockIdx.y;
  const h16* p = prev + (long)b * C * NPTS + n;
  float s = 0.f;
  for (int c = 0; c < C; ++c) {
    float v = (float)p[(long)c * NPTS];
    s += v * v;
  }
  xx[b * NPTS + n] = s;
}

// Top-8 of key(m) = 2*inner[n,m] - xx[m] (reference pair up to row constant).
__global__ void topk8(const float* __restrict__ inner, const float* __restrict__ xx,
                      int* __restrict__ idxout, int b)
{
  int n = blockIdx.x * blockDim.x + threadIdx.x;
  const float* row = inner + (long)n * NPTS;
  const float* xxb = xx + b * NPTS;
  float tv[KNN_K];
  int   ti[KNN_K];
#pragma unroll
  for (int q = 0; q < KNN_K; ++q) { tv[q] = -3.4e38f; ti[q] = 0; }
  for (int m = 0; m < NPTS; ++m) {
    float key = 2.f * row[m] - xxb[m];
    if (key > tv[KNN_K - 1]) {
      int p = KNN_K - 1;
      while (p > 0 && key > tv[p - 1]) { tv[p] = tv[p - 1]; ti[p] = ti[p - 1]; --p; }
      tv[p] = key; ti[p] = m;
    }
  }
  int* o = idxout + ((long)b * NPTS + n) * KNN_K;
#pragma unroll
  for (int q = 0; q < KNN_K; ++q) o[q] = ti[q];
}

// ===================== gumbel straight-through =====================
__global__ __launch_bounds__(256) void gumbel_hard(const float* __restrict__ logits,
                                                   const float* __restrict__ U,
                                                   float* __restrict__ out)
{
  const int o = blockIdx.x, b = blockIdx.y, tid = threadIdx.x;
  const float* L  = logits + ((long)b * 70 + o) * NPTS;
  const float* Ub = U + ((long)b * 70 + o) * NPTS;
  __shared__ float sv[256];
  __shared__ int   si[256];
  float bm = -3.4e38f; int bi = 0x7fffffff;
  for (int n = tid; n < NPTS; n += 256) {
    float z = L[n] - logf(-logf(Ub[n]));
    if (z > bm) { bm = z; bi = n; }
  }
  sv[tid] = bm; si[tid] = bi;
  __syncthreads();
  for (int s = 128; s > 0; s >>= 1) {
    if (tid < s) {
      float ov = sv[tid + s]; int oi = si[tid + s];
      if (ov > sv[tid] || (ov == sv[tid] && oi < si[tid])) { sv[tid] = ov; si[tid] = oi; }
    }
    __syncthreads();
  }
  float zmax = sv[0];
  int argi = si[0];
  __syncthreads();
  float ls = 0.f;
  for (int n = tid; n < NPTS; n += 256)
    ls += expf(L[n] - logf(-logf(Ub[n])) - zmax);
  sv[tid] = ls;
  __syncthreads();
  for (int s = 128; s > 0; s >>= 1) {
    if (tid < s) sv[tid] += sv[tid + s];
    __syncthreads();
  }
  float sarg = 1.0f / sv[0];
  float spec = (1.0f + sarg) - sarg;
  float* orow = out + ((long)o * NBATCH + b) * NPTS;
  for (int n = tid; n < NPTS; n += 256) orow[n] = (n == argi) ? spec : 0.0f;
}

// ===================== workspace layout (bytes) =====================
constexpr size_t MB = 1024u * 1024u;
constexpr size_t SC_P    = 0;                 // f16 P (<=16MB) | inner f32 | Z f16 (32MB) | logits f32
constexpr size_t SC_Q    = 16 * MB;
constexpr size_t SC_HMX  = 32 * MB;
constexpr size_t OFF_OUT0 = 48 * MB;             // f16 8x  64x2048
constexpr size_t OFF_OUT1 = OFF_OUT0 + 2 * MB;   // f16 8x 128x2048
constexpr size_t OFF_OUT2 = OFF_OUT1 + 4 * MB;   // f16 8x 256x2048
constexpr size_t OFF_OUT3 = OFF_OUT2 + 8 * MB;   // f16 8x 512x2048
constexpr size_t OFF_FEAT = OFF_OUT3 + 16 * MB;  // f16 8x1024x2048
constexpr size_t OFF_HM   = OFF_FEAT + 32 * MB;  // f16 8x 512x2048
constexpr size_t OFF_IDX  = OFF_HM + 16 * MB;    // i32 8x2048x8
constexpr size_t OFF_XX   = OFF_IDX + 512 * 1024;
constexpr size_t OFF_ST   = OFF_XX + 64 * 1024;  // 2048 f32 (sum | sumsq)
constexpr size_t OFF_WA1  = OFF_ST + 8 * 1024;
constexpr size_t OFF_WB1  = OFF_WA1 + 128 * 64 * 2;
constexpr size_t OFF_WA2  = OFF_WB1 + 128 * 64 * 2;
constexpr size_t OFF_WB2  = OFF_WA2 + 256 * 128 * 2;
constexpr size_t OFF_WA3  = OFF_WB2 + 256 * 128 * 2;
constexpr size_t OFF_WB3  = OFF_WA3 + 512 * 256 * 2;
constexpr size_t OFF_W4H  = OFF_WB3 + 512 * 256 * 2;          // 1024x960
constexpr size_t OFF_WM1H = OFF_W4H + 1024 * 960 * 2;         // 512x1024
constexpr size_t OFF_WM2H = OFF_WM1H + 512 * 1024 * 2;        // 80x512 (padded)

static inline void launch_gemm(hipStream_t s, int amode,
    const h16* A, long sa_i, long sa_k, long a_bs,
    const h16* Bm, long sb_k, long b_bs,
    void* Dp, int d_f16, long sd_i, long d_bs,
    int M, int Mrows, int Nd, int Kd, int nb, int accum)
{
  dim3 g(Nd / TN, (Mrows + TM - 1) / TM, nb), blk(128);
  if (amode == 0)
    gemm_wmma<0><<<g, blk, 0, s>>>(A, sa_i, sa_k, a_bs, Bm, sb_k, b_bs,
                                   Dp, d_f16, sd_i, d_bs, M, Mrows, Kd, accum);
  else
    gemm_wmma<1><<<g, blk, 0, s>>>(A, sa_i, sa_k, a_bs, Bm, sb_k, b_bs,
                                   Dp, d_f16, sd_i, d_bs, M, Mrows, Kd, accum);
}

extern "C" void kernel_launch(void* const* d_in, const int* in_sizes, int n_in,
                              void* d_out, int out_size, void* d_ws, size_t ws_size,
                              hipStream_t stream)
{
  (void)in_sizes; (void)n_in; (void)out_size; (void)ws_size;
  const float* x      = (const float*)d_in[0];
  const int*   sneigh = (const int*)d_in[1];
  const float* unoise = (const float*)d_in[2];
  const float* W0 = (const float*)d_in[3];
  const float* W1 = (const float*)d_in[4];
  const float* W2 = (const float*)d_in[5];
  const float* W3 = (const float*)d_in[6];
  const float* W4 = (const float*)d_in[7];
  const float* Wm1 = (const float*)d_in[8];
  const float* Wm2 = (const float*)d_in[9];
  float* out = (float*)d_out;

  char* ws = (char*)d_ws;
  h16*   P     = (h16*)(ws + SC_P);
  h16*   Q     = (h16*)(ws + SC_Q);
  h16*   HMX   = (h16*)(ws + SC_HMX);
  float* INNER = (float*)(ws + SC_P);   // time-shared with P
  h16*   Z     = (h16*)(ws + SC_P);     // time-shared pre-BN MLP buffer
  float* LOGITS = (float*)(ws + SC_P);  // time-shared
  h16* OUT0 = (h16*)(ws + OFF_OUT0);
  h16* OUT1 = (h16*)(ws + OFF_OUT1);
  h16* OUT2 = (h16*)(ws + OFF_OUT2);
  h16* OUT3 = (h16*)(ws + OFF_OUT3);
  h16* FEAT = (h16*)(ws + OFF_FEAT);
  h16* HM   = (h16*)(ws + OFF_HM);
  int* IDX  = (int*)(ws + OFF_IDX);
  float* XX = (float*)(ws + OFF_XX);
  float* SSUM = (float*)(ws + OFF_ST);
  float* SSQ  = SSUM + 1024;
  h16* WA[4] = { nullptr, (h16*)(ws + OFF_WA1), (h16*)(ws + OFF_WA2), (h16*)(ws + OFF_WA3) };
  h16* WB[4] = { nullptr, (h16*)(ws + OFF_WB1), (h16*)(ws + OFF_WB2), (h16*)(ws + OFF_WB3) };
  h16* W4H  = (h16*)(ws + OFF_W4H);
  h16* WM1H = (h16*)(ws + OFF_WM1H);
  h16* WM2H = (h16*)(ws + OFF_WM2H);

  const float inv_edge = 1.f / (float)(NBATCH * NPTS * KNN_K);
  const float inv_pt   = 1.f / (float)(NBATCH * NPTS);

  // ---- weight prep ----
  conv_split<<<(128 * 64 + 255) / 256, 256, 0, stream>>>(W1, WA[1], WB[1], 128, 64);
  conv_split<<<(256 * 128 + 255) / 256, 256, 0, stream>>>(W2, WA[2], WB[2], 256, 128);
  conv_split<<<(512 * 256 + 255) / 256, 256, 0, stream>>>(W3, WA[3], WB[3], 512, 256);
  conv_pad<<<(1024 * 960 + 255) / 256, 256, 0, stream>>>(W4, W4H, 1024, 960, 1024);
  conv_pad<<<(512 * 1024 + 255) / 256, 256, 0, stream>>>(Wm1, WM1H, 512, 1024, 512);
  conv_pad<<<(80 * 512 + 255) / 256, 256, 0, stream>>>(Wm2, WM2H, 70, 512, 80);

  // ---- layer 0 (C=3 input, O=64, idx = start_neighs) ----
  pq0_kernel<<<dim3(NPTS / 256, 64, NBATCH), 256, 0, stream>>>(x, W0, P, Q);
  zero_buf<<<8, 256, 0, stream>>>(SSUM, 2048);
  gather_max_stats<<<dim3(NPTS / 64, NBATCH), 64, 0, stream>>>(P, Q, sneigh, HMX, SSUM, SSQ, 64, 64);
  bn_lrelu<<<(NBATCH * 64 * NPTS) / 256, 256, 0, stream>>>(HMX, OUT0, SSUM, SSQ, inv_edge, 64);

  // ---- edge layers 1..3 ----
  const int Cs[4]     = { 0, 64, 128, 256 };
  h16*      prevs[4]  = { nullptr, OUT0, OUT1, OUT2 };
  h16*      outs[4]   = { nullptr, OUT1, OUT2, OUT3 };
  for (int l = 1; l <= 3; ++l) {
    const int C = Cs[l], O = 2 * C;
    const h16* prev = prevs[l];
    // KNN: xx, per-batch inner = prev^T * prev (WMMA), then top-8.
    sqnorm<<<dim3(NPTS / 256, NBATCH), 256, 0, stream>>>(prev, XX, C);
    for (int b = 0; b < NBATCH; ++b) {
      const h16* pb = prev + (long)b * C * NPTS;
      launch_gemm(stream, 1, pb, 1, NPTS, 0, pb, NPTS, 0,
                  INNER, 0, NPTS, 0, NPTS, NPTS, NPTS, C, 1, 0);
      topk8<<<NPTS / 128, 128, 0, stream>>>(INNER, XX, IDX, b);
    }
    // P = Wa*prev, Q = Wb*prev (batched WMMA GEMMs).
    launch_gemm(stream, 0, WA[l], C, 1, 0, prev, NPTS, (long)C * NPTS,
                P, 1, NPTS, (long)O * NPTS, O, O, NPTS, C, NBATCH, 0);
    launch_gemm(stream, 0, WB[l], C, 1, 0, prev, NPTS, (long)C * NPTS,
                Q, 1, NPTS, (long)O * NPTS, O, O, NPTS, C, NBATCH, 0);
    zero_buf<<<8, 256, 0, stream>>>(SSUM, 2048);
    gather_max_stats<<<dim3(NPTS / 64, NBATCH), O, 0, stream>>>(P, Q, IDX, HMX, SSUM, SSQ, O, 64);
    bn_lrelu<<<(NBATCH * O * NPTS) / 256, 256, 0, stream>>>(HMX, outs[l], SSUM, SSQ, inv_edge, O);
  }

  // ---- feat = lrelu(bn(W4 @ cat)) via 4 accumulating GEMMs ----
  {
    const int  coff[4] = { 0, 64, 192, 448 };
    const int  cg[4]   = { 64, 128, 256, 512 };
    h16*       bg[4]   = { OUT0, OUT1, OUT2, OUT3 };
    for (int g = 0; g < 4; ++g)
      launch_gemm(stream, 0, W4H + coff[g], 960, 1, 0, bg[g], NPTS, (long)cg[g] * NPTS,
                  Z, 1, NPTS, (long)1024 * NPTS, 1024, 1024, NPTS, cg[g], NBATCH, g > 0);
    zero_buf<<<8, 256, 0, stream>>>(SSUM, 2048);
    chan_stats<<<dim3(NPTS / 64, NBATCH), 1024, 0, stream>>>(Z, SSUM, SSQ, 1024, 64);
    bn_lrelu<<<(NBATCH * 1024 * NPTS) / 256, 256, 0, stream>>>(Z, FEAT, SSUM, SSQ, inv_pt, 1024);
  }

  // ---- hm = lrelu(bn(Wm1 @ feat)) ----
  launch_gemm(stream, 0, WM1H, 1024, 1, 0, FEAT, NPTS, (long)1024 * NPTS,
              Z, 1, NPTS, (long)512 * NPTS, 512, 512, NPTS, 1024, NBATCH, 0);
  zero_buf<<<8, 256, 0, stream>>>(SSUM, 2048);
  chan_stats<<<dim3(NPTS / 64, NBATCH), 512, 0, stream>>>(Z, SSUM, SSQ, 512, 64);
  bn_lrelu<<<(NBATCH * 512 * NPTS) / 256, 256, 0, stream>>>(Z, HM, SSUM, SSQ, inv_pt, 512);

  // ---- logits = Wm2 @ hm (M=70 padded to 80), f32 out ----
  launch_gemm(stream, 0, WM2H, 512, 1, 0, HM, NPTS, (long)512 * NPTS,
              LOGITS, 0, NPTS, (long)70 * NPTS, 70, 80, NPTS, 512, NBATCH, 0);

  // ---- gumbel straight-through hard one-hot ----
  gumbel_hard<<<dim3(70, NBATCH), 256, 0, stream>>>(LOGITS, unoise, out);
}
